// parallel_MLP_28819230556803
// MI455X (gfx1250) — matrix operand order repaired and checked by reading, not compiled, not run on hardware
//
#include <hip/hip_runtime.h>
#include <hip/hip_bf16.h>

// parallel MLP: out[t,n] = sum_w tanh(x[n]*w1[t,w]+b1[t,w])*w2[t,w] + b2[t]
// T=1024 tasks, N=2048 samples, W=50 hidden units.
//
// Strategy (MI455X / gfx1250, wave32):
//  - bottleneck is 105M tanh -> use native v_tanh_f32 (1 TRANS op each)
//  - pre-activation fma + f32->f16 cvt on VALU (co-executes with TRANS)
//  - the w-reduction runs on the WMMA pipe: A = 16x32 f16 tile of tanh
//    activations (16 samples x 32 w-chunk), B = w2 chunk replicated across
//    all 16 columns -> D[m,*] = dot(h_m, w2). W padded 50->64 with zeros.
//  - theta row staged zero-padded into LDS per block (8 waves share task t).

typedef __attribute__((ext_vector_type(16))) _Float16 v16h;
typedef __attribute__((ext_vector_type(8)))  float    v8f;

#define LAYERW 50
#define ROWLEN 151   // 3*W + 1
#define NKPAD  64    // W padded to 2 WMMA K-chunks

#if defined(__has_builtin) && __has_builtin(__builtin_amdgcn_tanhf)
#define FAST_TANH(x) __builtin_amdgcn_tanhf(x)
#else
__device__ __forceinline__ float fast_tanh_asm(float v) {
  float r;
  asm("v_tanh_f32 %0, %1" : "=v"(r) : "v"(v));
  return r;
}
#define FAST_TANH(x) fast_tanh_asm(x)
#endif

__global__ __launch_bounds__(256)
void parallel_mlp_wmma_kernel(const float* __restrict__ theta,
                              const float* __restrict__ x,
                              float* __restrict__ out,
                              int N) {
  __shared__ float s_w1[NKPAD];
  __shared__ float s_b1[NKPAD];
  __shared__ float s_w2[NKPAD];

  const int t   = blockIdx.y;
  const int tid = threadIdx.x;
  const float* __restrict__ row = theta + (size_t)t * ROWLEN;

  // Stage theta row into LDS, zero-padded from 50 to 64 per array.
  if (tid < NKPAD) {
    s_w1[tid] = (tid < LAYERW) ? row[tid] : 0.0f;
  } else if (tid < 2 * NKPAD) {
    const int i = tid - NKPAD;
    s_b1[i] = (i < LAYERW) ? row[LAYERW + i] : 0.0f;
  } else if (tid < 3 * NKPAD) {
    const int i = tid - 2 * NKPAD;
    s_w2[i] = (i < LAYERW) ? row[2 * LAYERW + i] : 0.0f;
  }
  __syncthreads();

  const float b2 = row[3 * LAYERW];   // wave-uniform -> scalar load

  const int wave = tid >> 5;
  const int lane = tid & 31;
  const int hi   = lane >> 4;         // 0: lanes 0-15, 1: lanes 16-31
  const int m    = lane & 15;         // sample row within the 16-row tile
  const int n0   = blockIdx.x * 128 + wave * 16;

  const float xv = x[n0 + m];

  v8f c = {};

  // Two K-chunks of 32 cover the padded 64 hidden units.
  #pragma unroll
  for (int wc = 0; wc < NKPAD; wc += 32) {
    v16h a, b;

    // A-matrix (16x32 f16): lane holds row M = m.
    //   elements 0..7  -> K = wc + hi*8 + e
    //   elements 8..15 -> K = wc + 16 + hi*8 + (e-8)
    #pragma unroll
    for (int e = 0; e < 8; ++e) {
      {
        const int k = wc + hi * 8 + e;
        const float h = FAST_TANH(fmaf(xv, s_w1[k], s_b1[k]));
        a[e] = (_Float16)h;
      }
      {
        const int k = wc + 16 + hi * 8 + e;
        const float h = FAST_TANH(fmaf(xv, s_w1[k], s_b1[k]));
        a[8 + e] = (_Float16)h;
      }
    }

    // B-matrix (32x16 f16): row K = wc + (e&8) + hi*8 + e (same K order as A), identical across columns.
    #pragma unroll
    for (int e = 0; e < 16; ++e) {
      b[e] = (_Float16)s_w2[wc + (e & 8) + hi * 8 + e];
    }

    // c += A x B  (f32 accumulate)
    c = __builtin_amdgcn_wmma_f32_16x16x32_f16(
        false, a, false, b, (short)0, c, false, false);
  }

  // D layout: lane holds rows M = r + hi*8 (r = VGPR index), every column
  // carries the same dot product. Lane with m==0 of each half writes 8 rows.
  if (m == 0) {
    float* __restrict__ o = out + (size_t)t * N + n0 + hi * 8;
    #pragma unroll
    for (int r = 0; r < 8; ++r) {
      o[r] = c[r] + b2;
    }
  }
}

extern "C" void kernel_launch(void* const* d_in, const int* in_sizes, int n_in,
                              void* d_out, int out_size, void* d_ws, size_t ws_size,
                              hipStream_t stream) {
  const float* theta = (const float*)d_in[0];
  const float* x     = (const float*)d_in[1];
  float*       out   = (float*)d_out;

  const int T = in_sizes[0] / ROWLEN;   // 1024 tasks
  const int N = in_sizes[1];            // 2048 samples

  // One block = one task x 128 samples (8 waves x 16-sample WMMA tiles).
  dim3 grid(N / 128, T);
  parallel_mlp_wmma_kernel<<<grid, 256, 0, stream>>>(theta, x, out, N);
}
